// GraphDirectionPredictionModel_1073741824486
// MI455X (gfx1250) — matrix operand order repaired and checked by reading, not compiled, not run on hardware
//
#include <hip/hip_runtime.h>
#include <hip/hip_bf16.h>

typedef __attribute__((ext_vector_type(16))) _Float16 v16h;
typedef __attribute__((ext_vector_type(8)))  float    v8f;

#define N_NODES 100000

// ---------------------------------------------------------------------------
// Degree / normalization kernels
// ---------------------------------------------------------------------------
__global__ void k_init_deg(float* __restrict__ deg, int n) {
    unsigned i = blockIdx.x * blockDim.x + threadIdx.x;
    if (i < (unsigned)n) deg[i] = 1.0f;   // self-loop contributes 1
}

__global__ void k_deg_accum(const int* __restrict__ dst, float* __restrict__ deg, int E) {
    unsigned e = blockIdx.x * blockDim.x + threadIdx.x;
    if (e < (unsigned)E) unsafeAtomicAdd(&deg[dst[e]], 1.0f);
}

__global__ void k_rsqrt_inplace(float* __restrict__ deg, int n) {
    unsigned i = blockIdx.x * blockDim.x + threadIdx.x;
    if (i < (unsigned)n) deg[i] = __frsqrt_rn(deg[i]);  // deg >= 1 always
}

// ---------------------------------------------------------------------------
// Pre-swizzle W[128 x N] (f32 row-major) into f16 B-fragment order:
//   Wf[(((nTile*4 + kk)*32 + lane)*16 + t)] = W[(kk*32 + (t<8 ? (lane>>4)*8 + t : 16 + (lane>>4)*8 + t-8))*N
//                                               + nTile*16 + (lane&15)]
// so each wave's per-step B fragment is one contiguous 32-byte v16h.
// ---------------------------------------------------------------------------
__global__ void k_prep_B(const float* __restrict__ W, _Float16* __restrict__ Wf, int N) {
    unsigned tid = blockIdx.x * blockDim.x + threadIdx.x;
    if (tid >= (unsigned)N * 128) return;
    const unsigned t     = tid & 15;
    const unsigned lane  = (tid >> 4) & 31;
    const unsigned kk    = (tid >> 9) & 3;
    const unsigned nTile = tid >> 11;
    const unsigned half  = lane >> 4;
    const unsigned l16   = lane & 15;
    const unsigned K   = kk * 32 + (t < 8 ? half * 8 + t : 16 + half * 8 + (t - 8));
    const unsigned col = nTile * 16 + l16;
    Wf[tid] = (_Float16)W[(size_t)K * N + col];
}

// ---------------------------------------------------------------------------
// GEMM: H[M x N] = act(A)[M x 128] * W[128 x N], fp32 in/out, f16 WMMA compute.
// One wave per 16x16 output tile; K=128 as 4 x (16x16x32) WMMA steps.
// B comes pre-swizzled (one v16h load per step). Optional fused ReLU on A.
// ---------------------------------------------------------------------------
template <bool RELU>
__global__ __launch_bounds__(128)
void k_gemm_wmma(const float* __restrict__ A, const _Float16* __restrict__ Wf,
                 float* __restrict__ H, int N) {
    const int lane = threadIdx.x & 31;
    const int wave = threadIdx.x >> 5;
    const int half = lane >> 4;    // 0: lanes 0-15, 1: lanes 16-31
    const int l16  = lane & 15;

    const int mBase    = blockIdx.x * 16;
    const int nTileIdx = blockIdx.y * 4 + wave;
    const int col      = nTileIdx * 16 + l16;

    const float* __restrict__ aRow = A + (size_t)(mBase + l16) * 128;
    const _Float16* __restrict__ wf = Wf + ((size_t)nTileIdx * 4 * 32 + lane) * 16;

    v8f c = {0.f, 0.f, 0.f, 0.f, 0.f, 0.f, 0.f, 0.f};

#pragma unroll
    for (int kk = 0; kk < 4; ++kk) {
        const int kbase = kk * 32;

        // ---- A fragment (16x32 f16): lane half 0 -> K {kbase+0..7, kbase+16..23}
        //                              lane half 1 -> K {kbase+8..15, kbase+24..31}
        const int ka0 = kbase + half * 8;
        const int ka1 = kbase + 16 + half * 8;
        float4 a0 = *reinterpret_cast<const float4*>(aRow + ka0);
        float4 a1 = *reinterpret_cast<const float4*>(aRow + ka0 + 4);
        float4 a2 = *reinterpret_cast<const float4*>(aRow + ka1);
        float4 a3 = *reinterpret_cast<const float4*>(aRow + ka1 + 4);
        if (RELU) {
            a0.x=fmaxf(a0.x,0.f); a0.y=fmaxf(a0.y,0.f); a0.z=fmaxf(a0.z,0.f); a0.w=fmaxf(a0.w,0.f);
            a1.x=fmaxf(a1.x,0.f); a1.y=fmaxf(a1.y,0.f); a1.z=fmaxf(a1.z,0.f); a1.w=fmaxf(a1.w,0.f);
            a2.x=fmaxf(a2.x,0.f); a2.y=fmaxf(a2.y,0.f); a2.z=fmaxf(a2.z,0.f); a2.w=fmaxf(a2.w,0.f);
            a3.x=fmaxf(a3.x,0.f); a3.y=fmaxf(a3.y,0.f); a3.z=fmaxf(a3.z,0.f); a3.w=fmaxf(a3.w,0.f);
        }
        v16h a;
        a[0]=(_Float16)a0.x; a[1]=(_Float16)a0.y; a[2]=(_Float16)a0.z; a[3]=(_Float16)a0.w;
        a[4]=(_Float16)a1.x; a[5]=(_Float16)a1.y; a[6]=(_Float16)a1.z; a[7]=(_Float16)a1.w;
        a[8]=(_Float16)a2.x; a[9]=(_Float16)a2.y; a[10]=(_Float16)a2.z; a[11]=(_Float16)a2.w;
        a[12]=(_Float16)a3.x; a[13]=(_Float16)a3.y; a[14]=(_Float16)a3.z; a[15]=(_Float16)a3.w;

        // ---- B fragment: contiguous pre-swizzled v16h (32 bytes)
        const v16h b = *reinterpret_cast<const v16h*>(wf + (size_t)kk * 32 * 16);

        c = __builtin_amdgcn_wmma_f32_16x16x32_f16(
                /*neg_a=*/false, a, /*neg_b=*/false, b,
                /*c_mod=*/(short)0, c, /*reuse_a=*/false, /*reuse_b=*/false);
    }

    // C/D layout: VGPR r, lanes 0-15 -> M=r, lanes 16-31 -> M=8+r
#pragma unroll
    for (int r = 0; r < 8; ++r)
        H[(size_t)(mBase + half * 8 + r) * N + col] = c[r];
}

// ---------------------------------------------------------------------------
// out[i][j] = bias[j] + h[i][j] * dinv[i]^2   (self-loop message + bias init)
// float4-vectorized streaming pass.
// ---------------------------------------------------------------------------
template <int D>
__global__ void k_init_self(const float* __restrict__ h, const float* __restrict__ bias,
                            const float* __restrict__ dinv, float* __restrict__ out, int n) {
    constexpr int P = D / 4;
    unsigned tid = blockIdx.x * blockDim.x + threadIdx.x;
    if (tid >= (unsigned)n * P) return;
    const unsigned i  = tid / P;
    const unsigned jv = tid % P;
    const float di = dinv[i];
    const float w  = di * di;
    const float4 hv = *reinterpret_cast<const float4*>(h + (size_t)tid * 4);
    const float4 bv = *reinterpret_cast<const float4*>(bias + jv * 4);
    float4 o;
    o.x = bv.x + hv.x * w;
    o.y = bv.y + hv.y * w;
    o.z = bv.z + hv.z * w;
    o.w = bv.w + hv.w * w;
    *reinterpret_cast<float4*>(out + (size_t)tid * 4) = o;
}

// ---------------------------------------------------------------------------
// Edge scatter: out[dst] += h[src] * (dinv[src]*dinv[dst]).
// One wave per edge; src/dst/norm forced wave-uniform via readfirstlane.
// Lane handles D/32 consecutive floats; hardware global_atomic_add_f32.
// ---------------------------------------------------------------------------
template <int D>
__global__ void k_edge_scatter(const int* __restrict__ src, const int* __restrict__ dst,
                               const float* __restrict__ dinv, const float* __restrict__ h,
                               float* __restrict__ out, int E) {
    constexpr int VPT = D / 32;  // floats per lane (4 for D=128, 2 for D=64)
    const unsigned tid = blockIdx.x * blockDim.x + threadIdx.x;
    const unsigned e = tid >> 5;         // wave-uniform edge id
    if (e >= (unsigned)E) return;        // uniform exit (whole wave)
    const int lane = threadIdx.x & 31;

    const int s = __builtin_amdgcn_readfirstlane(src[e]);
    const int d = __builtin_amdgcn_readfirstlane(dst[e]);
    const float norm = dinv[s] * dinv[d];

    const float* hp = h   + (size_t)s * D + lane * VPT;
    float*       op = out + (size_t)d * D + lane * VPT;

    if (VPT == 4) {
        const float4 hv = *reinterpret_cast<const float4*>(hp);
        unsafeAtomicAdd(op + 0, hv.x * norm);
        unsafeAtomicAdd(op + 1, hv.y * norm);
        unsafeAtomicAdd(op + 2, hv.z * norm);
        unsafeAtomicAdd(op + 3, hv.w * norm);
    } else {
        const float2 hv = *reinterpret_cast<const float2*>(hp);
        unsafeAtomicAdd(op + 0, hv.x * norm);
        unsafeAtomicAdd(op + 1, hv.y * norm);
    }
}

// ---------------------------------------------------------------------------
// Launch
// ---------------------------------------------------------------------------
extern "C" void kernel_launch(void* const* d_in, const int* in_sizes, int n_in,
                              void* d_out, int out_size, void* d_ws, size_t ws_size,
                              hipStream_t stream) {
    const float* x  = (const float*)d_in[0];
    const int*   ei = (const int*)d_in[1];
    const float* W1 = (const float*)d_in[2];
    const float* b1 = (const float*)d_in[3];
    const float* W2 = (const float*)d_in[4];
    const float* b2 = (const float*)d_in[5];
    float* out = (float*)d_out;

    const int n = N_NODES;
    const int E = in_sizes[1] / 2;
    const int* src = ei;
    const int* dst = ei + E;

    // Workspace layout (bytes):
    //   [0,      400000)   deg -> dinv (in place)
    //   [409600, +32768)   Wf1 : pre-swizzled f16 W1 fragments
    //   [442368, +16384)   Wf2 : pre-swizzled f16 W2 fragments
    //   [524288, +51.2M)   h1  (layer-1 GEMM out) ; later h3 (25.6M, reused)
    //   [524288+51.2M, +51.2M)  agg1 (layer-1 accumulator / h2)
    char* ws = (char*)d_ws;
    float*     deg  = (float*)ws;
    _Float16*  Wf1  = (_Float16*)(ws + 409600);
    _Float16*  Wf2  = (_Float16*)(ws + 442368);
    float*     h1   = (float*)(ws + 524288);
    float*     agg1 = (float*)(ws + 524288 + 51200000);
    float*     h3   = h1;  // reused after layer-1 scatter completes

    const int T = 256;

    // 1) degrees + symmetric norm
    k_init_deg<<<(n + T - 1) / T, T, 0, stream>>>(deg, n);
    k_deg_accum<<<(E + T - 1) / T, T, 0, stream>>>(dst, deg, E);
    k_rsqrt_inplace<<<(n + T - 1) / T, T, 0, stream>>>(deg, n);

    // 2) pre-swizzle both weight matrices into f16 fragment order
    k_prep_B<<<(128 * 128 + T - 1) / T, T, 0, stream>>>(W1, Wf1, 128);
    k_prep_B<<<(128 * 64 + T - 1) / T, T, 0, stream>>>(W2, Wf2, 64);

    // 3) h1 = x @ W1   [100000x128] * [128x128]
    k_gemm_wmma<false><<<dim3(n / 16, 2), 128, 0, stream>>>(x, Wf1, h1, 128);

    // 4) agg1 = b1 + self-loop msg; scatter edge messages
    k_init_self<128><<<((unsigned)n * 32 + T - 1) / T, T, 0, stream>>>(h1, b1, deg, agg1, n);
    k_edge_scatter<128><<<((unsigned)E * 32 + T - 1) / T, T, 0, stream>>>(src, dst, deg, h1, agg1, E);

    // 5) h3 = relu(agg1) @ W2   (ReLU fused into GEMM A-path)
    k_gemm_wmma<true><<<dim3(n / 16, 1), 128, 0, stream>>>(agg1, Wf2, h3, 64);

    // 6) out = b2 + self-loop msg + edge messages
    k_init_self<64><<<((unsigned)n * 16 + T - 1) / T, T, 0, stream>>>(h3, b2, deg, out, n);
    k_edge_scatter<64><<<((unsigned)E * 16 + T - 1) / T, T, 0, stream>>>(src, dst, deg, h3, out, E);
}
